// AttentionHead_2284922601686
// MI455X (gfx1250) — matrix operand order repaired and checked by reading, not compiled, not run on hardware
//
#include <hip/hip_runtime.h>

typedef __attribute__((ext_vector_type(16))) _Float16 v16h;
typedef __attribute__((ext_vector_type(8)))  float    v8f;
typedef int i32x4 __attribute__((vector_size(16)));

#define D_MODEL 1024
#define DK      64
#define SEQ     4096
#define NBATCH  4

// ---- gfx1250 async global->LDS copy (ASYNCcnt) with portable fallback ------
#if defined(__has_builtin)
#  if __has_builtin(__builtin_amdgcn_global_load_async_to_lds_b128)
#    define HAS_ASYNC_LDS 1
#  endif
#endif

__device__ __forceinline__ void copy16_to_lds(const _Float16* g, _Float16* l)
{
#ifdef HAS_ASYNC_LDS
    __builtin_amdgcn_global_load_async_to_lds_b128(
        (__attribute__((address_space(1))) i32x4*)g,
        (__attribute__((address_space(3))) i32x4*)l, 0, 0);
#else
    *(uint4*)l = *(const uint4*)g;
#endif
}

__device__ __forceinline__ void async_join()
{
#ifdef HAS_ASYNC_LDS
    asm volatile("s_wait_asynccnt 0x0" ::: "memory");
#endif
}

// ---------------------------------------------------------------------------
// Kernel 1: fused projections  Y = X @ W + b  -> f16 workspace.
// grid (128, 3): x = 128-row tile of flattened [B*S, D_MODEL], y selects q/k/v
// block 256 = 8 waves; each wave computes a 16(M) x 64(N) strip via WMMA.
// Global loads for tile kc+32 are software-pipelined against WMMA on tile kc.
// V output is stored TRANSPOSED ([b][d][s]) for the attention kernel.
// ---------------------------------------------------------------------------
__global__ __launch_bounds__(256)
void proj_kernel(const float* __restrict__ Q, const float* __restrict__ K,
                 const float* __restrict__ V,
                 const float* __restrict__ Wq, const float* __restrict__ Wk,
                 const float* __restrict__ Wv,
                 const float* __restrict__ bq, const float* __restrict__ bk,
                 const float* __restrict__ bv,
                 _Float16* __restrict__ qh, _Float16* __restrict__ kh,
                 _Float16* __restrict__ vt)
{
    const float* X; const float* W; const float* bias; _Float16* Y;
    if (blockIdx.y == 0)      { X = Q; W = Wq; bias = bq; Y = qh; }
    else if (blockIdx.y == 1) { X = K; W = Wk; bias = bk; Y = kh; }
    else                      { X = V; W = Wv; bias = bv; Y = vt; }
    const bool vtrans = (blockIdx.y == 2);

    __shared__ __align__(16) _Float16 sA[128 * 32];   // X tile, [row][k]
    __shared__ __align__(16) _Float16 sWt[64 * 32];   // W tile TRANSPOSED [n][k]

    const int tid  = threadIdx.x;
    const int wave = tid >> 5;
    const int lane = tid & 31;
    const int half = lane >> 4;   // 0: lanes 0-15, 1: lanes 16-31
    const int l16  = lane & 15;
    const long tile_m = (long)blockIdx.x * 128;

    // uniform-trip staged loads (4 x4-chunks of X, 2 of W per thread)
    auto load_tiles = [&](int kc, float4 (&x4)[4], float4 (&w4)[2]) {
        #pragma unroll
        for (int jj = 0; jj < 4; ++jj) {
            int j  = tid + jj * 256;
            int r  = j >> 3;
            int c4 = (j & 7) * 4;
            x4[jj] = *(const float4*)&X[(tile_m + r) * D_MODEL + kc + c4];
        }
        #pragma unroll
        for (int jj = 0; jj < 2; ++jj) {
            int j  = tid + jj * 256;
            int r  = j >> 4;          // k row 0..31
            int c4 = (j & 15) * 4;    // n col 0..60
            w4[jj] = *(const float4*)&W[(kc + r) * DK + c4];
        }
    };
    auto store_tiles = [&](const float4 (&x4)[4], const float4 (&w4)[2]) {
        #pragma unroll
        for (int jj = 0; jj < 4; ++jj) {
            int j  = tid + jj * 256;
            int r  = j >> 3;
            int c4 = (j & 7) * 4;
            _Float16* d = &sA[r * 32 + c4];
            d[0] = (_Float16)x4[jj].x; d[1] = (_Float16)x4[jj].y;
            d[2] = (_Float16)x4[jj].z; d[3] = (_Float16)x4[jj].w;
        }
        #pragma unroll
        for (int jj = 0; jj < 2; ++jj) {
            int j  = tid + jj * 256;
            int r  = j >> 4;
            int c4 = (j & 15) * 4;
            sWt[(c4 + 0) * 32 + r] = (_Float16)w4[jj].x;
            sWt[(c4 + 1) * 32 + r] = (_Float16)w4[jj].y;
            sWt[(c4 + 2) * 32 + r] = (_Float16)w4[jj].z;
            sWt[(c4 + 3) * 32 + r] = (_Float16)w4[jj].w;
        }
    };

    v8f acc[4] = {};
    float4 x4[4]; float4 w4[2];
    load_tiles(0, x4, w4);

    for (int kc = 0; kc < D_MODEL; kc += 32) {
        store_tiles(x4, w4);
        __syncthreads();

        // prefetch next tile into registers while WMMAs run
        if (kc + 32 < D_MODEL) load_tiles(kc + 32, x4, w4);

        // A fragment: lane m=l16; elem i -> K = half*16 + i  (contiguous)
        v16h a;
        {
            const int m = wave * 16 + l16;
            #pragma unroll
            for (int i = 0; i < 16; ++i) {
                int k = half * 16 + i;
                a[i] = sA[m * 32 + k];
            }
        }
        #pragma unroll
        for (int nt = 0; nt < 4; ++nt) {
            // B fragment: lane n=l16; elem i -> K = half*16 + i  (contiguous)
            v16h bfr;
            const int n = nt * 16 + l16;
            #pragma unroll
            for (int i = 0; i < 16; ++i)
                bfr[i] = sWt[n * 32 + half * 16 + i];
            acc[nt] = __builtin_amdgcn_wmma_f32_16x16x32_f16(
                false, a, false, bfr, (short)0, acc[nt], false, false);
        }
        __syncthreads();
    }

    // epilogue: + bias, f16, store (V transposed to [b][d][s])
    #pragma unroll
    for (int nt = 0; nt < 4; ++nt) {
        #pragma unroll
        for (int r = 0; r < 8; ++r) {
            int row = wave * 16 + half * 8 + r;   // C layout: VGPR r -> row r+half*8
            int col = nt * 16 + l16;
            float val = acc[nt][r] + bias[col];
            long grow = tile_m + row;
            if (vtrans) {
                long bb = grow >> 12;             // /SEQ
                long s  = grow & (SEQ - 1);
                Y[bb * (long)SEQ * DK + (long)col * SEQ + s] = (_Float16)val;
            } else {
                Y[grow * DK + col] = (_Float16)val;
            }
        }
    }
}

// ---------------------------------------------------------------------------
// Kernel 2: flash attention. grid (SEQ/64, B); block 128 = 4 waves.
// Double-buffered K/V tiles staged with async global->LDS b128 copies.
// ---------------------------------------------------------------------------
__global__ __launch_bounds__(128)
void attn_kernel(const _Float16* __restrict__ qh, const _Float16* __restrict__ kh,
                 const _Float16* __restrict__ vt, float* __restrict__ out)
{
    __shared__ __align__(16) _Float16 sK [2][64 * 64];  // [key][dim]
    __shared__ __align__(16) _Float16 sVt[2][64 * 64];  // [dim][key] (transposed)
    __shared__ __align__(16) _Float16 sP [64 * 64];     // [qrow][key]

    const int tid  = threadIdx.x;
    const int wave = tid >> 5;
    const int lane = tid & 31;
    const int half = lane >> 4;
    const int l16  = lane & 15;

    const int  b      = blockIdx.y;
    const long base   = (long)b * SEQ * DK;
    const long tile_m = (long)blockIdx.x * 64;

    const _Float16* q = qh + base;
    const _Float16* k = kh + base;
    const _Float16* v = vt + base;   // [DK][SEQ]

    // stage one 64-key tile into buffer bi (async if available)
    auto stage = [&](int kt, int bi) {
        // K tile: [key][dim] is 8 KB contiguous in memory
        const _Float16* src = k + (long)kt * DK;
        #pragma unroll
        for (int j = 0; j < 4; ++j) {
            int idx = tid + j * 128;                 // 512 chunks of 8 halves
            copy16_to_lds(src + idx * 8, &sK[bi][idx * 8]);
        }
        // V tile: sVt[d][key] <- vt[d*SEQ + kt + key]
        #pragma unroll
        for (int j = 0; j < 4; ++j) {
            int idx = tid + j * 128;
            int d   = idx >> 3;
            int k0  = (idx & 7) * 8;
            copy16_to_lds(v + (long)d * SEQ + kt + k0, &sVt[bi][d * 64 + k0]);
        }
    };

    // Q rows as two 16x32 A fragments (head dim 64), registers.
    v16h aq[2];
    {
        const long m = tile_m + wave * 16 + l16;
        #pragma unroll
        for (int c = 0; c < 2; ++c)
            #pragma unroll
            for (int i = 0; i < 16; ++i) {
                int d = c * 32 + half * 16 + i;
                aq[c][i] = q[m * DK + d];
            }
    }

    v8f   acc_o[4] = {};
    float m_i[8], l_i[8];
    #pragma unroll
    for (int r = 0; r < 8; ++r) { m_i[r] = -1e30f; l_i[r] = 0.0f; }

    stage(0, 0);
    async_join();
    __syncthreads();

    for (int kt = 0; kt < SEQ; kt += 64) {
        const int cur = (kt >> 6) & 1;
        if (kt + 64 < SEQ) stage(kt + 64, cur ^ 1);   // prefetch next tile

        // S = Q K^T  (16 x 64 per wave); contiguous B reads from sK
        v8f s_acc[4] = {};
        #pragma unroll
        for (int c = 0; c < 2; ++c) {
            #pragma unroll
            for (int nt = 0; nt < 4; ++nt) {
                v16h bf;
                const int key = nt * 16 + l16;
                const int d0  = c * 32 + half * 16;
                #pragma unroll
                for (int i = 0; i < 16; ++i)
                    bf[i] = sK[cur][key * 64 + d0 + i];
                s_acc[nt] = __builtin_amdgcn_wmma_f32_16x16x32_f16(
                    false, aq[c], false, bf, (short)0, s_acc[nt], false, false);
            }
        }

        // online softmax: reduce across the 16 lanes holding each row
        #pragma unroll
        for (int r = 0; r < 8; ++r) {
            float mx = -1e30f;
            #pragma unroll
            for (int nt = 0; nt < 4; ++nt) {
                s_acc[nt][r] *= 0.125f;              // 1/sqrt(64)
                mx = fmaxf(mx, s_acc[nt][r]);
            }
            #pragma unroll
            for (int off = 1; off < 16; off <<= 1)
                mx = fmaxf(mx, __shfl_xor(mx, off, 32));
            float m_new = fmaxf(m_i[r], mx);
            float alpha = __expf(m_i[r] - m_new);
            float rs = 0.0f;
            #pragma unroll
            for (int nt = 0; nt < 4; ++nt) {
                float p = __expf(s_acc[nt][r] - m_new);
                s_acc[nt][r] = p;
                rs += p;
            }
            #pragma unroll
            for (int off = 1; off < 16; off <<= 1)
                rs += __shfl_xor(rs, off, 32);
            l_i[r] = l_i[r] * alpha + rs;
            m_i[r] = m_new;
            #pragma unroll
            for (int nt = 0; nt < 4; ++nt)
                acc_o[nt][r] *= alpha;
        }

        // C-layout -> A-layout for P via wave-private LDS strip
        #pragma unroll
        for (int nt = 0; nt < 4; ++nt)
            #pragma unroll
            for (int r = 0; r < 8; ++r) {
                int row = wave * 16 + half * 8 + r;
                int col = nt * 16 + l16;
                sP[row * 64 + col] = (_Float16)s_acc[nt][r];
            }
        asm volatile("s_wait_dscnt 0" ::: "memory");  // wave-local LDS RAW

        // O += P V ; contiguous A reads from sP, contiguous B reads from sVt
        #pragma unroll
        for (int c = 0; c < 2; ++c) {
            v16h ap;
            const int m2 = wave * 16 + l16;
            #pragma unroll
            for (int i = 0; i < 16; ++i) {
                int key = c * 32 + half * 16 + i;
                ap[i] = sP[m2 * 64 + key];
            }
            #pragma unroll
            for (int nt = 0; nt < 4; ++nt) {
                v16h bv2;
                const int dcol = nt * 16 + l16;
                const int k0   = c * 32 + half * 16;
                #pragma unroll
                for (int i = 0; i < 16; ++i)
                    bv2[i] = sVt[cur][dcol * 64 + k0 + i];
                acc_o[nt] = __builtin_amdgcn_wmma_f32_16x16x32_f16(
                    false, ap, false, bv2, (short)0, acc_o[nt], false, false);
            }
        }

        async_join();      // this wave's prefetch landed
        __syncthreads();   // all waves done with cur + prefetch complete
    }

    // epilogue: O / l -> fp32 out [B,S,DK]
    #pragma unroll
    for (int nt = 0; nt < 4; ++nt)
        #pragma unroll
        for (int r = 0; r < 8; ++r) {
            long row = tile_m + wave * 16 + half * 8 + r;
            int  col = nt * 16 + l16;
            out[base + row * DK + col] = acc_o[nt][r] / l_i[r];
        }
}

// ---------------------------------------------------------------------------
extern "C" void kernel_launch(void* const* d_in, const int* in_sizes, int n_in,
                              void* d_out, int out_size, void* d_ws, size_t ws_size,
                              hipStream_t stream)
{
    const float* Q  = (const float*)d_in[0];
    const float* K  = (const float*)d_in[1];
    const float* V  = (const float*)d_in[2];
    const float* Wq = (const float*)d_in[3];
    const float* bq = (const float*)d_in[4];
    const float* Wk = (const float*)d_in[5];
    const float* bk = (const float*)d_in[6];
    const float* Wv = (const float*)d_in[7];
    const float* bv = (const float*)d_in[8];

    // f16 q/k in [B,S,DK]; v transposed [B,DK,S]. 3 x 2 MB workspace.
    _Float16* qh = (_Float16*)d_ws;
    _Float16* kh = qh + (size_t)NBATCH * SEQ * DK;
    _Float16* vt = kh + (size_t)NBATCH * SEQ * DK;

    proj_kernel<<<dim3((NBATCH * SEQ) / 128, 3), 256, 0, stream>>>(
        Q, K, V, Wq, Wk, Wv, bq, bk, bv, qh, kh, vt);

    attn_kernel<<<dim3(SEQ / 64, NBATCH), 128, 0, stream>>>(
        qh, kh, vt, (float*)d_out);
}
